// KimiK25Nvfp4MLAAttention_24026047054669
// MI455X (gfx1250) — matrix operand order repaired and checked by reading, not compiled, not run on hardware
//
#include <hip/hip_runtime.h>
#include <hip/hip_bf16.h>
#include <cstdint>
#include <cstddef>

// ---------------- problem constants ----------------
#define T_SEQ   2048
#define HID     7168
#define NH      16
#define QL      1536
#define KVL     512
#define D_NOPE  128
#define D_ROPE  64
#define D_V     128
#define QKD     192                     // D_NOPE + D_ROPE
#define QKV_A_N (QL + KVL + D_ROPE)     // 2112
#define QKV_PAD 2176                    // 2112 padded to multiple of 128
#define ATT_SCALE 0.07216878364870322f  // 192^-0.5

typedef __attribute__((ext_vector_type(16))) __bf16 bf16x16;
typedef __attribute__((ext_vector_type(8)))  float  f32x8;
typedef __attribute__((ext_vector_type(4)))  unsigned int u32x4;
typedef __attribute__((ext_vector_type(8)))  int i32x8;
typedef __attribute__((ext_vector_type(4)))  int i32x4;

union BFrag {
  bf16x16 v;
  uint4   q[2];
  unsigned short u[16];
};

__device__ __forceinline__ unsigned short f32_to_bf16(float f) {
  unsigned int u = __float_as_uint(f);
  u += 0x7FFFu + ((u >> 16) & 1u);   // round-to-nearest-even
  return (unsigned short)(u >> 16);
}

__device__ __forceinline__ f32x8 zero8() {
  f32x8 z = {0.f, 0.f, 0.f, 0.f, 0.f, 0.f, 0.f, 0.f};
  return z;
}

// ---------------- Tensor Data Mover: 2D tile load global->LDS ----------------
// D# per CDNA5 ISA ch8: group0 {count=1, lds_addr, global_addr, type=2},
// group1 {data_size=1(2B), tensor_dim0/1, tile_dim0/1, tensor_dim0_stride},
// groups 2/3 zero (2D tensor). All values uniform -> SGPR descriptor.
// This toolchain declares the 6-arg builtin:
//   (u32x4 g0, i32x8 g1, i32x4 g2, i32x4 g3, i32x8 gx, i32 cpol)
__device__ __forceinline__ void tdm_load_2d(
    unsigned lds_addr, const void* gaddr,
    unsigned tensor_d0, unsigned tensor_d1,
    unsigned tile_d0, unsigned tile_d1,
    unsigned long long stride0_elems)
{
  unsigned long long ga = (unsigned long long)(uintptr_t)gaddr;
  u32x4 g0;
  g0[0] = 1u;                                            // count=1 valid descriptor
  g0[1] = lds_addr;                                      // LDS byte address
  g0[2] = (unsigned)(ga & 0xFFFFFFFFu);                  // global_addr[31:0]
  g0[3] = (unsigned)((ga >> 32) & 0x1FFFFFFu) | (2u << 30); // global_addr[56:32] | type=2
  i32x8 g1;
  g1[0] = (int)(1u << 16);                               // data_size=1 (2 bytes)
  g1[1] = (int)((tensor_d0 & 0xFFFFu) << 16);            // tensor_dim0[15:0]
  g1[2] = (int)((tensor_d0 >> 16) | ((tensor_d1 & 0xFFFFu) << 16));
  g1[3] = (int)((tensor_d1 >> 16) | (tile_d0 << 16));    // tensor_dim1[31:16] | tile_dim0
  g1[4] = (int)(tile_d1 & 0xFFFFu);                      // tile_dim1 | tile_dim2=0
  g1[5] = (int)(stride0_elems & 0xFFFFFFFFu);            // tensor_dim0_stride[31:0]
  g1[6] = (int)((stride0_elems >> 32) & 0xFFFFu);        // stride[47:32] | dim1_stride lo = 0
  g1[7] = 0;
  i32x4 gz4 = {0, 0, 0, 0};
  i32x8 gz8 = {0, 0, 0, 0, 0, 0, 0, 0};
  __builtin_amdgcn_tensor_load_to_lds(g0, g1, gz4, gz4, gz8, 0);
}

// ---------------- fp32 -> bf16 conversion (vectorized x4) ----------------
__global__ __launch_bounds__(256) void cvt_f32_bf16_v4(
    const float4* __restrict__ in, ushort4* __restrict__ out, size_t n4) {
  size_t i = (size_t)blockIdx.x * 256 + threadIdx.x;
  const size_t stride = (size_t)gridDim.x * 256;
  for (; i < n4; i += stride) {
    float4 f = in[i];
    ushort4 o;
    o.x = f32_to_bf16(f.x); o.y = f32_to_bf16(f.y);
    o.z = f32_to_bf16(f.z); o.w = f32_to_bf16(f.w);
    out[i] = o;
  }
}

// ---------------- bf16 WMMA GEMM: C[M,Npad] = A[M,K] * B[Npad,K]^T ----------------
// block = 256 threads = 8 waves (4 M x 2 N); block tile 256x128; wave tile 64x64.
// Requires M%256==0, Npad%128==0, K%32==0.
__global__ __launch_bounds__(256) void gemm_bf16(
    const unsigned short* __restrict__ A,   // [M,K] bf16
    const unsigned short* __restrict__ B,   // [Npad,K] bf16
    float* __restrict__ C,                  // [M,Npad] f32
    int Npad, int K) {
  const int wave  = threadIdx.x >> 5;
  const int lane  = threadIdx.x & 31;
  const int lhalf = lane >> 4;
  const int lmod  = lane & 15;
  const int waveM = blockIdx.y * 256 + (wave >> 1) * 64;
  const int waveN = blockIdx.x * 128 + (wave & 1) * 64;

  f32x8 acc[4][4];
#pragma unroll
  for (int mt = 0; mt < 4; ++mt)
#pragma unroll
    for (int nt = 0; nt < 4; ++nt) acc[mt][nt] = zero8();

  const unsigned short* aRow[4];
  const unsigned short* bRow[4];
#pragma unroll
  for (int mt = 0; mt < 4; ++mt)
    aRow[mt] = A + (size_t)(waveM + mt * 16 + lmod) * K + lhalf * 16;
#pragma unroll
  for (int nt = 0; nt < 4; ++nt)
    bRow[nt] = B + (size_t)(waveN + nt * 16 + lmod) * K + lhalf * 16;

  for (int k = 0; k < K; k += 32) {
    BFrag a[4], b[4];
#pragma unroll
    for (int mt = 0; mt < 4; ++mt) {
      const unsigned short* p = aRow[mt] + k;
      a[mt].q[0] = *(const uint4*)p;          // K 0..7   (lanes>=16: 16..23)
      a[mt].q[1] = *(const uint4*)(p + 8);    // K 8..15  (lanes>=16: 24..31)
    }
#pragma unroll
    for (int nt = 0; nt < 4; ++nt) {
      const unsigned short* p = bRow[nt] + k;
      b[nt].q[0] = *(const uint4*)p;          // this half K 0..7
      b[nt].q[1] = *(const uint4*)(p + 8);    // this half K 8..15
    }
#pragma unroll
    for (int mt = 0; mt < 4; ++mt)
#pragma unroll
      for (int nt = 0; nt < 4; ++nt)
        acc[mt][nt] = __builtin_amdgcn_wmma_f32_16x16x32_bf16(
            false, a[mt].v, false, b[nt].v, (short)0, acc[mt][nt], false, false);
  }

#pragma unroll
  for (int mt = 0; mt < 4; ++mt)
#pragma unroll
    for (int nt = 0; nt < 4; ++nt) {
      const int col = waveN + nt * 16 + lmod;
#pragma unroll
      for (int r = 0; r < 8; ++r) {
        const int row = waveM + mt * 16 + lhalf * 8 + r;
        C[(size_t)row * Npad + col] = acc[mt][nt][r];
      }
    }
}

// ---------------- RMS norm (fp32 in, bf16 out), one block per row ----------------
__global__ __launch_bounds__(256) void rmsnorm_bf16(
    const float* __restrict__ in, int in_stride, int in_off,
    const float* __restrict__ w, unsigned short* __restrict__ out, int L) {
  const int row = blockIdx.x;
  const float* x = in + (size_t)row * in_stride + in_off;
  float ss = 0.f;
  for (int i = threadIdx.x; i < L; i += 256) { float v = x[i]; ss += v * v; }
#pragma unroll
  for (int m = 16; m >= 1; m >>= 1) ss += __shfl_xor(ss, m, 32);
  __shared__ float red[8];
  if ((threadIdx.x & 31) == 0) red[threadIdx.x >> 5] = ss;
  __syncthreads();
  float tot = 0.f;
#pragma unroll
  for (int i = 0; i < 8; ++i) tot += red[i];
  const float rn = rsqrtf(tot / (float)L + 1e-6f);
  for (int i = threadIdx.x; i < L; i += 256)
    out[(size_t)row * L + i] = f32_to_bf16(x[i] * rn * w[i]);
}

// ---------------- build Q[H,T,192], K[H,T,192], Vt[H,128,T] (bf16) + RoPE ----------------
__global__ __launch_bounds__(256) void prep_qkv(
    const int* __restrict__ positions,
    const float* __restrict__ qkva,   // [T, QKV_PAD] f32 (first 2112 valid)
    const float* __restrict__ qf,     // [T, 3072] f32
    const float* __restrict__ kvf,    // [T, 4096] f32
    unsigned short* __restrict__ Qh,  // [H, T, 192]
    unsigned short* __restrict__ Kh,  // [H, T, 192]
    unsigned short* __restrict__ Vt)  // [H, 128, T]
{
  const int t = blockIdx.x;
  const float pos = (float)positions[t];
  __shared__ float kpe[D_ROPE];
  if (threadIdx.x < D_ROPE / 2) {
    const int i = threadIdx.x;
    const float invf = __powf(10000.f, -((float)(2 * i)) / (float)D_ROPE);
    const float ang = pos * invf;
    const float c = __cosf(ang), s = __sinf(ang);
    const float x1 = qkva[(size_t)t * QKV_PAD + QL + KVL + 2 * i];
    const float x2 = qkva[(size_t)t * QKV_PAD + QL + KVL + 2 * i + 1];
    kpe[2 * i]     = x1 * c - x2 * s;
    kpe[2 * i + 1] = x2 * c + x1 * s;
  }
  __syncthreads();
  for (int idx = threadIdx.x; idx < NH * QKD; idx += 256) {
    const int h = idx / QKD, d = idx % QKD;
    float qv, kv;
    if (d < D_NOPE) {
      qv = qf[(size_t)t * (NH * QKD) + h * QKD + d];
      kv = kvf[(size_t)t * (NH * (D_NOPE + D_V)) + h * (D_NOPE + D_V) + d];
    } else {
      const int dd = d - D_NOPE;
      const int i = dd >> 1;
      const float invf = __powf(10000.f, -((float)(2 * i)) / (float)D_ROPE);
      const float ang = pos * invf;
      const float c = __cosf(ang), s = __sinf(ang);
      const float x1 = qf[(size_t)t * (NH * QKD) + h * QKD + D_NOPE + 2 * i];
      const float x2 = qf[(size_t)t * (NH * QKD) + h * QKD + D_NOPE + 2 * i + 1];
      qv = (dd & 1) ? (x2 * c + x1 * s) : (x1 * c - x2 * s);
      kv = kpe[dd];
    }
    Qh[((size_t)h * T_SEQ + t) * QKD + d] = f32_to_bf16(qv);
    Kh[((size_t)h * T_SEQ + t) * QKD + d] = f32_to_bf16(kv);
  }
  for (int idx = threadIdx.x; idx < NH * D_V; idx += 256) {
    const int h = idx / D_V, d = idx % D_V;
    Vt[((size_t)h * D_V + d) * T_SEQ + t] =
        f32_to_bf16(kvf[(size_t)t * (NH * (D_NOPE + D_V)) + h * (D_NOPE + D_V) + D_NOPE + d]);
  }
}

// ---------------- flash attention (causal), bf16 WMMA + double-buffered TDM K/V ----------------
// block = 128 threads = 4 waves; each wave owns a 16-row query tile.
// Wave 0 DMAs the shared 32-key K/V tiles into LDS via the Tensor Data Mover,
// double-buffered so tile i+1 streams in while tile i is consumed.
// TENSORcnt decrements in order -> s_wait_tensorcnt(2) = oldest tile complete.
__global__ __launch_bounds__(128) void flash_attn(
    const unsigned short* __restrict__ Q,   // [H, T, 192]
    const unsigned short* __restrict__ K,   // [H, T, 192]
    const unsigned short* __restrict__ Vt,  // [H, 128, T]
    unsigned short* __restrict__ attn)      // [T, H*128] bf16
{
  const int h     = blockIdx.y;
  const int wave  = threadIdx.x >> 5;
  const int lane  = threadIdx.x & 31;
  const int lhalf = lane >> 4;
  const int lmod  = lane & 15;
  const int qb0   = blockIdx.x * 64;
  const int qb    = qb0 + wave * 16;

  __shared__ __align__(16) unsigned short ldsK[2][32 * QKD];   // 24 KB: [buf][key][feat]
  __shared__ __align__(16) unsigned short ldsV[2][D_V * 32];   // 16 KB: [buf][vdim][key]
  __shared__ __align__(16) unsigned short pbuf[4][16 * 32];    //  4 KB: per-wave P staging
  unsigned short* pl = &pbuf[wave][0];

  const unsigned short* Qh = Q  + (size_t)h * T_SEQ * QKD;
  const unsigned short* Kh = K  + (size_t)h * T_SEQ * QKD;
  const unsigned short* Vh = Vt + (size_t)h * D_V * T_SEQ;

  // preload Q fragments (6 x 32-wide slices of the 192 feature dim)
  BFrag qfr[6];
  {
    const int row = qb + lmod;
#pragma unroll
    for (int f = 0; f < 6; ++f) {
      const unsigned short* p = Qh + (size_t)row * QKD + f * 32 + lhalf * 16;
      qfr[f].q[0] = *(const uint4*)p;
      qfr[f].q[1] = *(const uint4*)(p + 8);
    }
  }

  f32x8 o[8];
#pragma unroll
  for (int nt = 0; nt < 8; ++nt) o[nt] = zero8();
  float m_run[8], l_run[8];
#pragma unroll
  for (int r = 0; r < 8; ++r) { m_run[r] = -1e30f; l_run[r] = 0.f; }

  const int nTiles = (qb0 + 64) / 32;   // block-uniform causal extent

  if (wave == 0) {
    // prime buffer 0 with tile 0
    tdm_load_2d((unsigned)(uintptr_t)&ldsK[0][0], Kh, QKD, T_SEQ, QKD, 32, QKD);
    tdm_load_2d((unsigned)(uintptr_t)&ldsV[0][0], Vh, T_SEQ, D_V, 32, D_V, T_SEQ);
  }

  for (int it = 0; it < nTiles; ++it) {
    const int kb = it * 32;
    const int p  = it & 1;
    if (wave == 0) {
      if (it + 1 < nTiles) {
        const int kn = kb + 32;
        tdm_load_2d((unsigned)(uintptr_t)&ldsK[1 - p][0], Kh + (size_t)kn * QKD,
                    QKD, (unsigned)(T_SEQ - kn), QKD, 32, QKD);
        tdm_load_2d((unsigned)(uintptr_t)&ldsV[1 - p][0], Vh + kn,
                    (unsigned)(T_SEQ - kn), D_V, 32, D_V, T_SEQ);
        __builtin_amdgcn_s_wait_tensorcnt(2);   // oldest 2 ops (this tile) done
      } else {
        __builtin_amdgcn_s_wait_tensorcnt(0);
      }
    }
    __syncthreads();

    f32x8 s0 = zero8(), s1 = zero8();
#pragma unroll
    for (int f = 0; f < 6; ++f) {
      BFrag b0, b1;
      const unsigned short* p0 = &ldsK[p][(size_t)lmod * QKD + f * 32 + lhalf * 16];
      const unsigned short* p1 = &ldsK[p][(size_t)(16 + lmod) * QKD + f * 32 + lhalf * 16];
      b0.q[0] = *(const uint4*)p0; b0.q[1] = *(const uint4*)(p0 + 8);
      b1.q[0] = *(const uint4*)p1; b1.q[1] = *(const uint4*)(p1 + 8);
      s0 = __builtin_amdgcn_wmma_f32_16x16x32_bf16(false, qfr[f].v, false, b0.v, (short)0, s0, false, false);
      s1 = __builtin_amdgcn_wmma_f32_16x16x32_bf16(false, qfr[f].v, false, b1.v, (short)0, s1, false, false);
    }

    float alpha[8];
#pragma unroll
    for (int r = 0; r < 8; ++r) {
      const int row  = qb + lhalf * 8 + r;
      const int key0 = kb + lmod;
      const int key1 = kb + 16 + lmod;
      float v0 = s0[r] * ATT_SCALE; if (key0 > row) v0 = -1e30f;
      float v1 = s1[r] * ATT_SCALE; if (key1 > row) v1 = -1e30f;
      float mx = fmaxf(v0, v1);
#pragma unroll
      for (int msk = 1; msk < 16; msk <<= 1) mx = fmaxf(mx, __shfl_xor(mx, msk, 32));
      const float mnew = fmaxf(m_run[r], mx);
      const float p0 = __expf(v0 - mnew);
      const float p1 = __expf(v1 - mnew);
      float ls = p0 + p1;
#pragma unroll
      for (int msk = 1; msk < 16; msk <<= 1) ls += __shfl_xor(ls, msk, 32);
      const float a = __expf(m_run[r] - mnew);
      l_run[r] = l_run[r] * a + ls;
      m_run[r] = mnew;
      alpha[r] = a;
      pl[(lhalf * 8 + r) * 32 + lmod]      = f32_to_bf16(p0);
      pl[(lhalf * 8 + r) * 32 + 16 + lmod] = f32_to_bf16(p1);
    }
#pragma unroll
    for (int nt = 0; nt < 8; ++nt)
#pragma unroll
      for (int r = 0; r < 8; ++r) o[nt][r] = o[nt][r] * alpha[r];

    // reload P in A-fragment layout from LDS
    BFrag pf;
    const unsigned short* pp = pl + lmod * 32 + lhalf * 16;
    pf.q[0] = *(const uint4*)pp;
    pf.q[1] = *(const uint4*)(pp + 8);

#pragma unroll
    for (int nt = 0; nt < 8; ++nt) {
      BFrag vf;
      const unsigned short* vp = &ldsV[p][(size_t)(nt * 16 + lmod) * 32 + lhalf * 16];
      vf.q[0] = *(const uint4*)vp;
      vf.q[1] = *(const uint4*)(vp + 8);
      o[nt] = __builtin_amdgcn_wmma_f32_16x16x32_bf16(false, pf.v, false, vf.v, (short)0, o[nt], false, false);
    }
    __syncthreads();   // all waves done with this buffer before it is re-filled
  }

#pragma unroll
  for (int nt = 0; nt < 8; ++nt)
#pragma unroll
    for (int r = 0; r < 8; ++r) {
      const int row = qb + lhalf * 8 + r;
      const int col = h * D_V + nt * 16 + lmod;
      attn[(size_t)row * (NH * D_V) + col] = f32_to_bf16(o[nt][r] / l_run[r]);
    }
}

// ---------------- host launcher ----------------
extern "C" void kernel_launch(void* const* d_in, const int* in_sizes, int n_in,
                              void* d_out, int out_size, void* d_ws, size_t ws_size,
                              hipStream_t stream) {
  (void)in_sizes; (void)n_in; (void)out_size; (void)ws_size;

  const int*   positions = (const int*)d_in[0];
  const float* hs        = (const float*)d_in[1];
  const float* w_qkv_a   = (const float*)d_in[2];
  const float* q_ln_w    = (const float*)d_in[3];
  const float* w_qb      = (const float*)d_in[4];
  const float* kv_ln_w   = (const float*)d_in[5];
  const float* w_kvb     = (const float*)d_in[6];
  const float* w_o       = (const float*)d_in[7];
  float*       out       = (float*)d_out;

  char* ws = (char*)d_ws;
  size_t off = 0;
  auto alloc = [&](size_t bytes) -> void* {
    void* p = ws + off;
    off = (off + bytes + 255) & ~(size_t)255;
    return p;
  };

  unsigned short* hs_b    = (unsigned short*)alloc((size_t)T_SEQ * HID * 2);
  unsigned short* wqkva_b = (unsigned short*)alloc((size_t)QKV_PAD * HID * 2);  // rows >=2112 unused
  unsigned short* wqb_b   = (unsigned short*)alloc((size_t)(NH * QKD) * QL * 2);
  unsigned short* wkvb_b  = (unsigned short*)alloc((size_t)(NH * (D_NOPE + D_V)) * KVL * 2);
  unsigned short* wo_b    = (unsigned short*)alloc((size_t)HID * (NH * D_V) * 2);
  float*          qkva_f  = (float*)alloc((size_t)T_SEQ * QKV_PAD * 4);
  unsigned short* qc_b    = (unsigned short*)alloc((size_t)T_SEQ * QL * 2);
  unsigned short* kvc_b   = (unsigned short*)alloc((size_t)T_SEQ * KVL * 2);
  float*          q_f     = (float*)alloc((size_t)T_SEQ * NH * QKD * 4);
  float*          kv_f    = (float*)alloc((size_t)T_SEQ * NH * (D_NOPE + D_V) * 4);
  unsigned short* Qh      = (unsigned short*)alloc((size_t)NH * T_SEQ * QKD * 2);
  unsigned short* Kh      = (unsigned short*)alloc((size_t)NH * T_SEQ * QKD * 2);
  unsigned short* Vt      = (unsigned short*)alloc((size_t)NH * D_V * T_SEQ * 2);
  unsigned short* attn_b  = (unsigned short*)alloc((size_t)T_SEQ * NH * D_V * 2);

  auto cvt = [&](const float* src, unsigned short* dst, size_t n) {
    size_t n4 = n / 4;
    int grid = (int)((n4 + 255) / 256);
    if (grid > 4096) grid = 4096;
    cvt_f32_bf16_v4<<<grid, 256, 0, stream>>>((const float4*)src, (ushort4*)dst, n4);
  };
  cvt(hs,      hs_b,    (size_t)T_SEQ * HID);
  cvt(w_qkv_a, wqkva_b, (size_t)QKV_A_N * HID);
  cvt(w_qb,    wqb_b,   (size_t)(NH * QKD) * QL);
  cvt(w_kvb,   wkvb_b,  (size_t)(NH * (D_NOPE + D_V)) * KVL);
  cvt(w_o,     wo_b,    (size_t)HID * (NH * D_V));

  // qkv_a = hs @ w_qkv_a^T   [2048 x 2176(pad)]
  gemm_bf16<<<dim3(QKV_PAD / 128, T_SEQ / 256), 256, 0, stream>>>(
      hs_b, wqkva_b, qkva_f, QKV_PAD, HID);

  rmsnorm_bf16<<<T_SEQ, 256, 0, stream>>>(qkva_f, QKV_PAD, 0,  q_ln_w,  qc_b,  QL);
  rmsnorm_bf16<<<T_SEQ, 256, 0, stream>>>(qkva_f, QKV_PAD, QL, kv_ln_w, kvc_b, KVL);

  // q = q_c @ w_qb^T   [2048 x 3072]
  gemm_bf16<<<dim3((NH * QKD) / 128, T_SEQ / 256), 256, 0, stream>>>(
      qc_b, wqb_b, q_f, NH * QKD, QL);
  // kv = kv_c @ w_kvb^T [2048 x 4096]
  gemm_bf16<<<dim3((NH * (D_NOPE + D_V)) / 128, T_SEQ / 256), 256, 0, stream>>>(
      kvc_b, wkvb_b, kv_f, NH * (D_NOPE + D_V), KVL);

  prep_qkv<<<T_SEQ, 256, 0, stream>>>(positions, qkva_f, q_f, kv_f, Qh, Kh, Vt);

  flash_attn<<<dim3(T_SEQ / 64, NH), 128, 0, stream>>>(Qh, Kh, Vt, attn_b);

  // out = attn @ w_o^T  [2048 x 7168]
  gemm_bf16<<<dim3(HID / 128, T_SEQ / 256), 256, 0, stream>>>(
      attn_b, wo_b, out, HID, NH * D_V);
}
